// HetGTCN_mean_76682346102823
// MI455X (gfx1250) — matrix operand order repaired and checked by reading, not compiled, not run on hardware
//
#include <hip/hip_runtime.h>
#include <hip/hip_bf16.h>

// ---------- CDNA5 WMMA types ----------
typedef __attribute__((ext_vector_type(16))) __bf16         v16bf;
typedef __attribute__((ext_vector_type(8)))  float          v8f;
typedef __attribute__((ext_vector_type(8)))  unsigned short v8us;
typedef __attribute__((ext_vector_type(2)))  __bf16         v2bf;

__device__ __forceinline__ unsigned short f2bf(float f) {
    // round-to-nearest-even f32 -> bf16 (scalar fallback)
    unsigned int u = __float_as_uint(f);
    unsigned int r = u + 0x7FFFu + ((u >> 16) & 1u);
    return (unsigned short)(r >> 16);
}

__device__ __forceinline__ unsigned int pk_bf16(float a, float b) {
#if __has_builtin(__builtin_amdgcn_cvt_pk_bf16_f32)
    v2bf r = __builtin_amdgcn_cvt_pk_bf16_f32(a, b);
    return __builtin_bit_cast(unsigned int, r);
#else
    return ((unsigned int)f2bf(b) << 16) | (unsigned int)f2bf(a);
#endif
}

__device__ __forceinline__ void atomAddF32(float* p, float v) {
    __hip_atomic_fetch_add(p, v, __ATOMIC_RELAXED, __HIP_MEMORY_SCOPE_AGENT);
}

// ---------------------------------------------------------------------------
// Bulk f32 -> bf16 conversion (n elements, n % 8 == 0). One-time pass so the
// GEMM inner loop carries zero conversion VALU work.
// ---------------------------------------------------------------------------
__global__ __launch_bounds__(256)
void cvt_f32_to_bf16(const float* __restrict__ x, unsigned short* __restrict__ y,
                     long long n)
{
    const long long i = ((long long)blockIdx.x * blockDim.x + threadIdx.x) * 8;
    if (i >= n) return;
    const float4 f0 = *(const float4*)(x + i);
    const float4 f1 = *(const float4*)(x + i + 4);
    uint4 o;
    o.x = pk_bf16(f0.x, f0.y);
    o.y = pk_bf16(f0.z, f0.w);
    o.z = pk_bf16(f1.x, f1.y);
    o.w = pk_bf16(f1.z, f1.w);
    *(uint4*)(y + i) = o;
}

// ---------------------------------------------------------------------------
// Dense GEMM: Out[M,N] = act(Xb[M,K](bf16) @ W[K,N](f32) + bias[N]) via
// v_wmma_f32_16x16x32_bf16. Block = 256 threads = 8 waves.
// Block tile: 128 rows x 64 cols; each wave: 16 rows x 64 cols (4 WMMA tiles,
// one shared A fragment per K-step -> 2 global b128 + 8 ds b128 + 4 WMMA).
// W tile staged transposed in LDS (bf16, stride K+8 against bank conflicts).
// Requires: K % 32 == 0, K <= 256, N % 64 == 0.
// ---------------------------------------------------------------------------
#define LDS_KS 264   // 256 + 8 pad

__global__ __launch_bounds__(256)
void gemm_bias_act(const unsigned short* __restrict__ Xb, const float* __restrict__ W,
                   const float* __restrict__ bias, float* __restrict__ Out,
                   int M, int K, int N, int applyRelu)
{
    __shared__ __attribute__((aligned(16))) unsigned short ldsW[64 * LDS_KS];

    const int n_base  = blockIdx.y * 64;
    const int m_block = blockIdx.x * 128;
    const int tid     = threadIdx.x;

    // stage W[:, n_base:n_base+64) transposed into LDS as bf16: ldsW[n*LDS_KS+k]
    {
        const int tn   = tid & 63;        // column 0..63
        const int kset = tid >> 6;        // 0..3
        const int kcnt = K >> 2;          // K/4 rows per thread
        const int kst  = kset * kcnt;
        for (int k = kst; k < kst + kcnt; ++k)
            ldsW[tn * LDS_KS + k] = f2bf(W[(size_t)k * N + n_base + tn]);
    }
    __syncthreads();

    const int wave   = tid >> 5;
    const int lane   = tid & 31;
    const int lrow   = lane & 15;
    const int lhi    = lane >> 4;         // 0: lanes 0-15, 1: lanes 16-31
    const int m_base = m_block + wave * 16;

    int m  = m_base + lrow;
    int mc = (m < M) ? m : (M - 1);       // clamp: EXEC must stay all-1s for WMMA
    const unsigned short* xrow = Xb + (size_t)mc * K;

    v8f acc[4] = {};
    for (int k0 = 0; k0 < K; k0 += 32) {
        // ---- A fragment (16x32 bf16): elems 0..7 -> K=k0+8*lhi+j,
        //                               elems 8..15 -> K=k0+16+8*lhi+j
        const int ka = k0 + lhi * 8;
        struct { v8us lo, hi; } aa;
        aa.lo = *(const v8us*)(xrow + ka);
        aa.hi = *(const v8us*)(xrow + ka + 16);
        const v16bf a = __builtin_bit_cast(v16bf, aa);

        // ---- 4 B fragments + 4 WMMAs; B elems 0..7 -> K=k0+8*lhi+j, 8..15 -> K=k0+16+8*lhi+j, col=16*t+lrow
        const int kb = k0 + lhi * 8;
        #pragma unroll
        for (int t = 0; t < 4; ++t) {
            const unsigned short* bp = &ldsW[(t * 16 + lrow) * LDS_KS + kb];
            struct { v8us lo, hi; } bb;
            bb.lo = *(const v8us*)(bp);
            bb.hi = *(const v8us*)(bp + 16);
            const v16bf b = __builtin_bit_cast(v16bf, bb);
            acc[t] = __builtin_amdgcn_wmma_f32_16x16x32_bf16(
                         false, a, false, b, (short)0, acc[t], false, false);
        }
    }

    // D layout: VGPR r, lanes 0-15 -> M=r, lanes 16-31 -> M=r+8; N = lane%16
    if (m_block + 128 <= M) {
        // fast path: whole block tile in range, unguarded stores
        #pragma unroll
        for (int t = 0; t < 4; ++t) {
            const int n  = n_base + t * 16 + lrow;
            const float bv = bias[n];
            #pragma unroll
            for (int r = 0; r < 8; ++r) {
                const int mm = m_base + r + 8 * lhi;
                float v = acc[t][r] + bv;
                if (applyRelu) v = fmaxf(v, 0.0f);
                Out[(size_t)mm * N + n] = v;
            }
        }
    } else {
        #pragma unroll
        for (int t = 0; t < 4; ++t) {
            const int n  = n_base + t * 16 + lrow;
            const float bv = bias[n];
            #pragma unroll
            for (int r = 0; r < 8; ++r) {
                const int mm = m_base + r + 8 * lhi;
                if (mm < M) {
                    float v = acc[t][r] + bv;
                    if (applyRelu) v = fmaxf(v, 0.0f);
                    Out[(size_t)mm * N + n] = v;
                }
            }
        }
    }
}

// ---------------------------------------------------------------------------
// SpMM scatter: acc[row[e], :] += scale * val[e] * h[col[e], :], D = 256.
// 4 edges per 256-thread block; 64 lanes per edge, float4 per lane.
// ---------------------------------------------------------------------------
__global__ __launch_bounds__(256)
void spmm_scatter(const int* __restrict__ rowi, const int* __restrict__ coli,
                  const float* __restrict__ val, const float* __restrict__ h,
                  float* __restrict__ acc, int nE, float scale)
{
    const int e = blockIdx.x * 4 + (threadIdx.x >> 6);
    if (e >= nE) return;
    const int f  = (threadIdx.x & 63) * 4;
    const int r  = rowi[e];
    const int c  = coli[e];
    const float v = val[e] * scale;

    const float4 hv = *(const float4*)(h + (size_t)c * 256 + f);
    float* dst = acc + (size_t)r * 256 + f;
    atomAddF32(dst + 0, v * hv.x);
    atomAddF32(dst + 1, v * hv.y);
    atomAddF32(dst + 2, v * hv.z);
    atomAddF32(dst + 3, v * hv.w);
}

// ---------------------------------------------------------------------------
// acc[i, :] = scale * (d1[i] + d2[i]) * x[i, :]   (d2 may be null), D = 256
// ---------------------------------------------------------------------------
__global__ __launch_bounds__(256)
void init_acc(const float* __restrict__ x, const float* __restrict__ d1,
              const float* __restrict__ d2, float* __restrict__ acc,
              int nRows, float scale)
{
    const long long i     = (long long)blockIdx.x * blockDim.x + threadIdx.x;
    const long long total = (long long)nRows * 64;   // float4s per matrix
    if (i >= total) return;
    const int r = (int)(i >> 6);
    float s = d1[r];
    if (d2) s += d2[r];
    s *= scale;
    const float4 xv = ((const float4*)x)[i];
    float4 o;
    o.x = s * xv.x; o.y = s * xv.y; o.z = s * xv.z; o.w = s * xv.w;
    ((float4*)acc)[i] = o;
}

// ---------------------------------------------------------------------------
extern "C" void kernel_launch(void* const* d_in, const int* in_sizes, int n_in,
                              void* d_out, int out_size, void* d_ws, size_t ws_size,
                              hipStream_t stream)
{
    (void)n_in; (void)out_size; (void)ws_size;

    const float* x_P    = (const float*)d_in[0];
    const float* x_A    = (const float*)d_in[1];
    const float* x_S    = (const float*)d_in[2];
    const int*   pa_row = (const int*)d_in[3];
    const int*   pa_col = (const int*)d_in[4];
    const float* pa_val = (const float*)d_in[5];
    const int*   ps_row = (const int*)d_in[6];
    const int*   ps_col = (const int*)d_in[7];
    const float* ps_val = (const float*)d_in[8];
    const int*   ap_row = (const int*)d_in[9];
    const int*   ap_col = (const int*)d_in[10];
    const float* ap_val = (const float*)d_in[11];
    const int*   sp_row = (const int*)d_in[12];
    const int*   sp_col = (const int*)d_in[13];
    const float* sp_val = (const float*)d_in[14];
    const float* diag_PA = (const float*)d_in[15];
    const float* diag_PS = (const float*)d_in[16];
    const float* diag_AP = (const float*)d_in[17];
    const float* diag_SP = (const float*)d_in[18];
    const float* W1_P = (const float*)d_in[19];
    const float* b1_P = (const float*)d_in[20];
    const float* W1_A = (const float*)d_in[21];
    const float* b1_A = (const float*)d_in[22];
    const float* W1_S = (const float*)d_in[23];
    const float* b1_S = (const float*)d_in[24];
    const float* W2   = (const float*)d_in[25];
    const float* b2   = (const float*)d_in[26];
    float* out = (float*)d_out;

    const int N_HID = in_sizes[20];            // 256
    const int D_IN  = in_sizes[19] / N_HID;    // 256
    const int N_OUT = in_sizes[26];            // 64
    const int nP    = in_sizes[0] / D_IN;      // 100000
    const int nA    = in_sizes[1] / D_IN;      // 150000
    const int nS    = in_sizes[2] / D_IN;      // 1000
    const int ePA   = in_sizes[3];             // 800000
    const int ePS   = in_sizes[6];             // 100000

    const size_t szP = (size_t)nP * N_HID;
    const size_t szA = (size_t)nA * N_HID;
    const size_t szS = (size_t)nS * N_HID;

    // workspace layout: f32 region then bf16 region
    float* ws = (float*)d_ws;
    float* xP = ws;
    float* xA = xP + szP;
    float* xS = xA + szA;
    float* hP = xS + szS;
    float* hA = hP + szP;
    float* hS = hA + szA;
    unsigned short* bP = (unsigned short*)(hS + szS);   // also reused for hP(bf16)
    unsigned short* bA = bP + szP;
    unsigned short* bS = bA + szA;

    auto cdiv  = [](long long a, long long b) { return (int)((a + b - 1) / b); };

    // 1) input projections: x* = relu(x* @ W1_* + b1_*)  (A pre-converted to bf16)
    {
        dim3 blk(256);
        cvt_f32_to_bf16<<<cdiv(szP / 8, 256), blk, 0, stream>>>(x_P, bP, (long long)szP);
        cvt_f32_to_bf16<<<cdiv(szA / 8, 256), blk, 0, stream>>>(x_A, bA, (long long)szA);
        cvt_f32_to_bf16<<<cdiv(szS / 8, 256), blk, 0, stream>>>(x_S, bS, (long long)szS);
        dim3 gP(cdiv(nP, 128), N_HID / 64);
        dim3 gA(cdiv(nA, 128), N_HID / 64);
        dim3 gS(cdiv(nS, 128), N_HID / 64);
        gemm_bias_act<<<gP, blk, 0, stream>>>(bP, W1_P, b1_P, xP, nP, D_IN, N_HID, 1);
        gemm_bias_act<<<gA, blk, 0, stream>>>(bA, W1_A, b1_A, xA, nA, D_IN, N_HID, 1);
        gemm_bias_act<<<gS, blk, 0, stream>>>(bS, W1_S, b1_S, xS, nS, D_IN, N_HID, 1);
    }

    // 2) hop loop
    const float* hA_cur = xA;   // h = x at hop 0
    const float* hS_cur = xS;
    for (int hop = 0; hop < 3; ++hop) {
        // hP = 0.5*(spmm_pa(hA) + dPA*xP + spmm_ps(hS) + dPS*xP)
        init_acc<<<cdiv((long long)nP * 64, 256), 256, 0, stream>>>(xP, diag_PA, diag_PS, hP, nP, 0.5f);
        spmm_scatter<<<cdiv(ePA, 4), 256, 0, stream>>>(pa_row, pa_col, pa_val, hA_cur, hP, ePA, 0.5f);
        spmm_scatter<<<cdiv(ePS, 4), 256, 0, stream>>>(ps_row, ps_col, ps_val, hS_cur, hP, ePS, 0.5f);
        // hA = spmm_ap(hP) + dAP*xA
        init_acc<<<cdiv((long long)nA * 64, 256), 256, 0, stream>>>(xA, diag_AP, nullptr, hA, nA, 1.0f);
        spmm_scatter<<<cdiv(ePA, 4), 256, 0, stream>>>(ap_row, ap_col, ap_val, hP, hA, ePA, 1.0f);
        // hS = spmm_sp(hP) + dSP*xS
        init_acc<<<cdiv((long long)nS * 64, 256), 256, 0, stream>>>(xS, diag_SP, nullptr, hS, nS, 1.0f);
        spmm_scatter<<<cdiv(ePS, 4), 256, 0, stream>>>(sp_row, sp_col, sp_val, hP, hS, ePS, 1.0f);
        hA_cur = hA;
        hS_cur = hS;
    }

    // 3) out = hP @ W2 + b2   (hP converted to bf16 into bP)
    {
        dim3 blk(256);
        cvt_f32_to_bf16<<<cdiv(szP / 8, 256), blk, 0, stream>>>(hP, bP, (long long)szP);
        dim3 g(cdiv(nP, 128), N_OUT / 64);
        gemm_bias_act<<<g, blk, 0, stream>>>(bP, W2, b2, out, nP, N_HID, N_OUT, 0);
    }
}
